// COIN_27573690040582
// MI455X (gfx1250) — matrix operand order repaired and checked by reading, not compiled, not run on hardware
//
#include <hip/hip_runtime.h>
#include <stdint.h>

typedef __attribute__((ext_vector_type(16))) __bf16 v16bf;
typedef __attribute__((ext_vector_type(8)))  __bf16 v8bf;
typedef __attribute__((ext_vector_type(4)))  __bf16 v4bf;
typedef __attribute__((ext_vector_type(8)))  float  v8f;
typedef __attribute__((ext_vector_type(2)))  float  v2f;
typedef __attribute__((ext_vector_type(4)))  unsigned int v4u;
typedef __attribute__((ext_vector_type(8)))  int    v8i;
typedef __attribute__((ext_vector_type(4)))  int    v4i;

constexpr int   NR    = 16384;   // rows / Laplacian dim
constexpr int   INF   = 512;     // in_features
constexpr int   HIDD  = 256;     // hidden
constexpr int   NOUT  = 16;      // out_features
constexpr int   STEPS = 16;
constexpr float SG2   = 0.1f;

// A-fragment for 16-bit 16x32 A: lane holds row (lane&15); elements 0..7 are
// K = kb8..kb8+7 and elements 8..15 are K = kb8+16..kb8+23 (kb8 = (lane>>4)*8).
static __device__ __forceinline__ v16bf ldA16(const __bf16* p) {
  v8bf lo = *(const v8bf*)(p);
  v8bf hi = *(const v8bf*)(p + 16);
  return __builtin_shufflevector(lo, hi, 0,1,2,3,4,5,6,7,8,9,10,11,12,13,14,15);
}

static __device__ __forceinline__ v8f wmma_bf16(v16bf a, v16bf b, v8f c) {
  return __builtin_amdgcn_wmma_f32_16x16x32_bf16(false, a, false, b, (short)0, c,
                                                 false, false);
}

// ---- Tensor Data Mover: 2-D tile load, D# per CDNA5 ISA ch.8 (groups 0/1) ----
// data_size = 3 (8-byte units). All arguments are wave-uniform -> SGPR groups.
static __device__ __forceinline__ void tdm_load2d(
    uint64_t gaddr, unsigned ldsaddr,
    unsigned tile_d0, unsigned tile_d1,          // in 8B units / rows
    unsigned tensor_d0, unsigned tensor_d1,      // in 8B units / rows
    unsigned stride0)                            // row stride, 8B units
{
  v4u g0;
  g0[0] = 1u;                                         // count=1, user D#
  g0[1] = ldsaddr;                                    // LDS byte address
  g0[2] = (unsigned)gaddr;                            // global addr [31:0]
  g0[3] = (unsigned)((gaddr >> 32) & 0x01ffffffu)     // global addr [56:32]
          | (2u << 30);                               // type = 2 (image)
  v8i g1;
  g1[0] = (int)(3u << 16);                            // data_size = 8B
  g1[1] = (int)(tensor_d0 << 16);                     // tensor_dim0 [15:0]
  g1[2] = (int)((tensor_d0 >> 16) | (tensor_d1 << 16));
  g1[3] = (int)((tensor_d1 >> 16) | (tile_d0 << 16)); // tile_dim0
  g1[4] = (int)(tile_d1 & 0xffffu);                   // tile_dim1
  g1[5] = (int)stride0;                               // dim0 stride [31:0]
  g1[6] = 0;
  g1[7] = 0;
  v4i gz4 = {0, 0, 0, 0};                             // groups 2/3: 2-D tensor
  v8i gz8 = {0, 0, 0, 0, 0, 0, 0, 0};
  __builtin_amdgcn_tensor_load_to_lds(g0, g1, gz4, gz4, gz8, 0);
}

static __device__ __forceinline__ unsigned lds_off(const void* p) {
  // generic pointer to LDS: low 32 bits are the LDS address (ISA aperture rule)
  return (unsigned)(uintptr_t)p;
}

// ---------------- bulk fp32 -> bf16 convert, 4 elem / thread ----------------
__global__ void COIN_cvt4(const float4* __restrict__ src, v4bf* __restrict__ dst,
                          long n4) {
  long i = (long)blockIdx.x * blockDim.x + threadIdx.x;
  if (i >= n4) return;
  float4 v = src[i];
  v4bf o;
  o[0] = (__bf16)v.x; o[1] = (__bf16)v.y; o[2] = (__bf16)v.z; o[3] = (__bf16)v.w;
  dst[i] = o;
}

// ------------- transpose + convert: dst[c*K + k] = src[k*C + c] -------------
__global__ void COIN_tconv(const float* __restrict__ src, __bf16* __restrict__ dst,
                           int K, int C) {
  int id = blockIdx.x * blockDim.x + threadIdx.x;
  if (id >= K * C) return;
  int k = id / C, c = id - k * C;
  dst[(size_t)c * K + k] = (__bf16)src[id];
}

// --------------------------- fused MLP (bf16 WMMA) --------------------------
__global__ __launch_bounds__(128) void COIN_mlp_bf16(
    const __bf16* __restrict__ Xbf,  const __bf16* __restrict__ W1cm,
    const float*  __restrict__ b1,   const __bf16* __restrict__ W2cm,
    const float*  __restrict__ b2,   const __bf16* __restrict__ Wscm,
    const float*  __restrict__ bs,   float* __restrict__ urm,
    __bf16* __restrict__ ucm)
{
  __shared__ __bf16 h_lds[16][HIDD];      // 8 KB h tile (A of GEMM2)
  __shared__ float  red[4][16][16];       // 4 KB cross-wave reduction
  const int lane = threadIdx.x & 31;
  const int wave = threadIdx.x >> 5;
  const int r0   = blockIdx.x * 16;
  const int row  = lane & 15;
  const int col  = lane & 15;
  const int hi   = lane >> 4;
  const int kb8  = hi * 8;
  const int kb16 = hi * 16;

  const __bf16* xrow = Xbf + (size_t)(r0 + row) * INF + kb8;

  const v8f z = {0.f,0.f,0.f,0.f,0.f,0.f,0.f,0.f};
  v8f acc[4] = {z, z, z, z};
  for (int k0 = 0; k0 < INF; k0 += 32) {
    v16bf a = ldA16(xrow + k0);
#pragma unroll
    for (int t = 0; t < 4; ++t) {
      const __bf16* bp = W1cm + (size_t)(wave * 64 + t * 16 + col) * INF + k0 + kb8;
      acc[t] = wmma_bf16(a, ldA16(bp), acc[t]);
    }
  }
#pragma unroll
  for (int t = 0; t < 4; ++t) {
    int cg = wave * 64 + t * 16 + col;
    float bias = b1[cg];
#pragma unroll
    for (int v = 0; v < 8; ++v) {
      float x = acc[t][v] + bias;
      float e = x > 0.f ? x : (__expf(x) - 1.f);    // elu
      h_lds[v + 8 * hi][cg] = (__bf16)e;
    }
  }
  __syncthreads();

  v8f a2 = z;
  for (int kk = 0; kk < 128; kk += 32) {            // X @ Ws slice (K=512/4)
    int k0 = wave * 128 + kk;
    v16bf a = ldA16(Xbf + (size_t)(r0 + row) * INF + k0 + kb8);
    v16bf b = ldA16(Wscm + (size_t)col * INF + k0 + kb8);
    a2 = wmma_bf16(a, b, a2);
  }
  for (int kk = 0; kk < 64; kk += 32) {             // h @ W2 slice (K=256/4)
    int k0 = wave * 64 + kk;
    v16bf a = ldA16(&h_lds[row][k0 + kb8]);
    v16bf b = ldA16(W2cm + (size_t)col * HIDD + k0 + kb8);
    a2 = wmma_bf16(a, b, a2);
  }
#pragma unroll
  for (int v = 0; v < 8; ++v) red[wave][v + 8 * hi][col] = a2[v];
  __syncthreads();

  if (wave == 0) {
    float bias = bs[col] + b2[col];
#pragma unroll
    for (int v = 0; v < 8; ++v) {
      int m = v + 8 * hi;
      float val = red[0][m][col] + red[1][m][col] + red[2][m][col] +
                  red[3][m][col] + bias;
      urm[(size_t)(r0 + m) * NOUT + col] = val;              // fp32 master
      ucm[(size_t)col * NR + (r0 + m)]   = (__bf16)val;      // bf16 col-major
    }
  }
}

// ---------------- one diffusion step: out = u - sg2 * (L @ u) ---------------
// 256 blocks x 4 waves; block owns 64 rows. L and u chunks staged to LDS by
// the Tensor Data Mover, 4 buffers, 3 stages in flight. Inner loop is pure
// ds_load + v_wmma (no global loads, no loadcnt stalls).
__global__ __launch_bounds__(128) void COIN_diffuse_bf16(
    const __bf16* __restrict__ Lbf, const float* __restrict__ urm_in,
    const __bf16* __restrict__ ucm_in, float* __restrict__ urm_out,
    __bf16* __restrict__ ucm_out)
{
  constexpr int CH  = 128;          // K-chunk (bf16 elements)
  constexpr int NCH = NR / CH;      // 128 chunks
  __shared__ __align__(256) __bf16 Ltile[4][64][CH];   // 64 KB
  __shared__ __align__(256) __bf16 Utile[4][NOUT][CH]; // 16 KB

  const int lane = threadIdx.x & 31;
  const int wave = threadIdx.x >> 5;
  const int rblk = blockIdx.x * 64;
  const int row  = lane & 15;
  const int col  = lane & 15;
  const int hi   = lane >> 4;
  const int kb8  = hi * 8;
  const int kb16 = hi * 16;

  const uint64_t gL = (uint64_t)(uintptr_t)(Lbf + (size_t)rblk * NR);
  const uint64_t gU = (uint64_t)(uintptr_t)ucm_in;

  if (wave == 0) {
#pragma unroll
    for (int s = 0; s < 3; ++s) {
      // L chunk: 64 rows x 256 B, row stride 32768 B (= 4096 x 8B)
      tdm_load2d(gL + (uint64_t)s * (CH * 2), lds_off(&Ltile[s][0][0]),
                 32u, 64u, 4096u, (unsigned)NR, 4096u);
      // u chunk: 16 cols x 256 B (col-major), col stride 32768 B
      tdm_load2d(gU + (uint64_t)s * (CH * 2), lds_off(&Utile[s][0][0]),
                 32u, (unsigned)NOUT, 4096u, (unsigned)NOUT, 4096u);
    }
  }

  v8f acc = {0.f,0.f,0.f,0.f,0.f,0.f,0.f,0.f};
  for (int i = 0; i < NCH; ++i) {
    if (wave == 0) {
      if (i + 2 < NCH)      __builtin_amdgcn_s_wait_tensorcnt(4); // stage i done
      else if (i + 1 < NCH) __builtin_amdgcn_s_wait_tensorcnt(2);
      else                  __builtin_amdgcn_s_wait_tensorcnt(0);
    }
    __syncthreads();   // chunk i visible; all waves done with buffer (i+3)&3
    if (wave == 0 && i + 3 < NCH) {
      const int s = i + 3;
      tdm_load2d(gL + (uint64_t)s * (CH * 2), lds_off(&Ltile[s & 3][0][0]),
                 32u, 64u, 4096u, (unsigned)NR, 4096u);
      tdm_load2d(gU + (uint64_t)s * (CH * 2), lds_off(&Utile[s & 3][0][0]),
                 32u, (unsigned)NOUT, 4096u, (unsigned)NOUT, 4096u);
    }
    const __bf16* at = &Ltile[i & 3][wave * 16 + row][kb8];
    const __bf16* bt = &Utile[i & 3][col][kb8];
#pragma unroll
    for (int kk = 0; kk < CH; kk += 32) {
      v16bf a = ldA16(at + kk);
      v16bf b = ldA16(bt + kk);
      acc = wmma_bf16(a, b, acc);
    }
    __syncthreads();
  }

#pragma unroll
  for (int v = 0; v < 8; ++v) {
    int r = rblk + wave * 16 + v + 8 * hi;
    float val = urm_in[(size_t)r * NOUT + col] - SG2 * acc[v];
    urm_out[(size_t)r * NOUT + col] = val;
    ucm_out[(size_t)col * NR + r]   = (__bf16)val;
  }
}

// ------------------- fallback path (small workspace): fp32 ------------------
__global__ void COIN_mlp_scalar(
    const float* __restrict__ X,  const float* __restrict__ W1,
    const float* __restrict__ b1, const float* __restrict__ W2,
    const float* __restrict__ b2, const float* __restrict__ Ws,
    const float* __restrict__ bs, float* __restrict__ ucm)
{
  __shared__ float xs[INF];
  __shared__ float hs[HIDD];
  const int r = blockIdx.x;
  for (int k = threadIdx.x; k < INF; k += blockDim.x)
    xs[k] = X[(size_t)r * INF + k];
  __syncthreads();
  const int c = threadIdx.x;
  float acc = b1[c];
  for (int k = 0; k < INF; ++k) acc = fmaf(xs[k], W1[(size_t)k * HIDD + c], acc);
  hs[c] = acc > 0.f ? acc : (__expf(acc) - 1.f);
  __syncthreads();
  if (c < NOUT) {
    float o = bs[c] + b2[c];
    for (int k = 0; k < INF;  ++k) o = fmaf(xs[k], Ws[(size_t)k * NOUT + c], o);
    for (int k = 0; k < HIDD; ++k) o = fmaf(hs[k], W2[(size_t)k * NOUT + c], o);
    ucm[(size_t)c * NR + r] = o;
  }
}

__global__ __launch_bounds__(128) void COIN_diffuse_f32(
    const float* __restrict__ Lp, const float* __restrict__ uin,
    float* __restrict__ uout, float* __restrict__ dout)
{
  const int lane = threadIdx.x & 31;
  const int wave = threadIdx.x >> 5;
  const int r0   = blockIdx.x * 64 + wave * 16;
  const int row  = lane & 15;
  const int col  = lane & 15;
  const int hi   = lane >> 4;
  const float* ap = Lp  + (size_t)(r0 + row) * NR + hi * 2;
  const float* bp = uin + (size_t)col * NR + hi * 2;
  v8f acc = {0.f,0.f,0.f,0.f,0.f,0.f,0.f,0.f};
#pragma unroll 8
  for (int k0 = 0; k0 < NR; k0 += 4) {
    v2f a = *(const v2f*)(ap + k0);
    v2f b = *(const v2f*)(bp + k0);
    acc = __builtin_amdgcn_wmma_f32_16x16x4_f32(false, a, false, b, (short)0, acc,
                                                false, false);
  }
#pragma unroll
  for (int v = 0; v < 8; ++v) {
    int rr = r0 + v + 8 * hi;
    float val = uin[(size_t)col * NR + rr] - SG2 * acc[v];
    uout[(size_t)col * NR + rr] = val;
    if (dout) dout[(size_t)rr * NOUT + col] = val;
  }
}

extern "C" void kernel_launch(void* const* d_in, const int* in_sizes, int n_in,
                              void* d_out, int out_size, void* d_ws, size_t ws_size,
                              hipStream_t stream) {
  const float* X  = (const float*)d_in[0];
  const float* Lp = (const float*)d_in[1];
  const float* W1 = (const float*)d_in[2];
  const float* b1 = (const float*)d_in[3];
  const float* W2 = (const float*)d_in[4];
  const float* b2 = (const float*)d_in[5];
  const float* Ws = (const float*)d_in[6];
  const float* bs = (const float*)d_in[7];
  float* out = (float*)d_out;

  size_t off = 0;
  auto take = [&](size_t bytes) -> void* {
    void* p = (char*)d_ws + off;
    off += (bytes + (size_t)255) & ~(size_t)255;
    return p;
  };
  __bf16* Lbf  = (__bf16*)take((size_t)NR * NR * 2);      // 512 MB
  __bf16* Xbf  = (__bf16*)take((size_t)NR * INF * 2);     // 16 MB
  __bf16* W1cm = (__bf16*)take((size_t)INF * HIDD * 2);
  __bf16* W2cm = (__bf16*)take((size_t)HIDD * NOUT * 2);
  __bf16* Wscm = (__bf16*)take((size_t)INF * NOUT * 2);
  float*  urm0 = (float*)take((size_t)NR * NOUT * 4);
  float*  urm1 = (float*)take((size_t)NR * NOUT * 4);
  __bf16* ucm0 = (__bf16*)take((size_t)NR * NOUT * 2);
  __bf16* ucm1 = (__bf16*)take((size_t)NR * NOUT * 2);

  if (off <= ws_size) {
    // ---- primary path: bf16 L stream (halves HBM traffic vs fp32) ----
    long nL4 = (long)NR * NR / 4;
    COIN_cvt4<<<dim3((unsigned)(nL4 / 256)), dim3(256), 0, stream>>>(
        (const float4*)Lp, (v4bf*)Lbf, nL4);
    long nX4 = (long)NR * INF / 4;
    COIN_cvt4<<<dim3((unsigned)(nX4 / 256)), dim3(256), 0, stream>>>(
        (const float4*)X, (v4bf*)Xbf, nX4);
    COIN_tconv<<<dim3((INF * HIDD) / 256), dim3(256), 0, stream>>>(W1, W1cm, INF, HIDD);
    COIN_tconv<<<dim3((HIDD * NOUT) / 256), dim3(256), 0, stream>>>(W2, W2cm, HIDD, NOUT);
    COIN_tconv<<<dim3((INF * NOUT) / 256), dim3(256), 0, stream>>>(Ws, Wscm, INF, NOUT);

    COIN_mlp_bf16<<<dim3(NR / 16), dim3(128), 0, stream>>>(
        Xbf, W1cm, b1, W2cm, b2, Wscm, bs, urm0, ucm0);

    float*  rin = urm0;  __bf16* cin = ucm0;
    float*  rot = urm1;  __bf16* cot = ucm1;
    for (int s = 0; s < STEPS; ++s) {
      float* ro = (s == STEPS - 1) ? out : rot;
      COIN_diffuse_bf16<<<dim3(NR / 64), dim3(128), 0, stream>>>(Lbf, rin, cin, ro, cot);
      float*  tr = rin; rin = rot; rot = tr;
      __bf16* tc = cin; cin = cot; cot = tc;
    }
  } else {
    // ---- fallback: fp32 WMMA directly on L, ~2 MB scratch ----
    float* ua = (float*)d_ws;
    float* ub = ua + (size_t)NR * NOUT;
    COIN_mlp_scalar<<<dim3(NR), dim3(256), 0, stream>>>(X, W1, b1, W2, b2, Ws, bs, ua);
    float* in = ua;
    float* o  = ub;
    for (int s = 0; s < STEPS; ++s) {
      COIN_diffuse_f32<<<dim3(NR / 64), dim3(128), 0, stream>>>(
          Lp, in, o, (s == STEPS - 1) ? out : nullptr);
      float* t = in; in = o; o = t;
    }
  }
}
